// DDSP_3968549782201
// MI455X (gfx1250) — compile-verified, matrix operand order checked by reading, not run on hardware
//
#include <hip/hip_runtime.h>

typedef _Float16 f16;
typedef __attribute__((ext_vector_type(16))) _Float16 v16h;
typedef __attribute__((ext_vector_type(8)))  _Float16 v8h;
typedef __attribute__((ext_vector_type(8)))  float    v8f;
typedef __attribute__((ext_vector_type(4)))  unsigned int u32x4;
typedef __attribute__((ext_vector_type(8)))  unsigned int u32x8;

union V16 { v16h v; v8h h[2]; };

#define TOKN   4000      // B*F
#define FFR    1000
#define NBATCH 4
#define TSAMP  160000
#define BLK    160
#define REVN   16000
#define PI2F   6.28318530718f

static __device__ __forceinline__ float sigf(float x) { return 1.f / (1.f + __expf(-x)); }
static __device__ __forceinline__ float scalef(float x) {
  // 2*sigmoid(x)^ln(10) + 1e-7
  return 2.f * powf(sigf(x), 2.30258509299f) + 1e-7f;
}

// ---------------- weight prep ----------------
__global__ void k_cvt(const float* __restrict__ src, f16* __restrict__ dst,
                      int N, int K, int Np, int Kp) {
  int i = blockIdx.x * 256 + threadIdx.x;
  if (i >= Np * Kp) return;
  int n = i / Kp, k = i % Kp;
  dst[i] = (n < N && k < K) ? (f16)src[n * K + k] : (f16)0.f;
}

__global__ void k_padbias(const float* __restrict__ src, float* __restrict__ dst, int N, int Np) {
  int i = blockIdx.x * 256 + threadIdx.x;
  if (i < Np) dst[i] = (i < N) ? src[i] : 0.f;
}

// ---------------- mlp_in layer 1 (in_dim==1) + LN + leaky ----------------
__global__ void __launch_bounds__(512)
k_mlp_l1(const float* __restrict__ pitch, const float* __restrict__ loud,
         const float* __restrict__ W0, const float* __restrict__ b0,
         const float* __restrict__ g, const float* __restrict__ be,
         f16* __restrict__ out) {
  int tok = blockIdx.x, c = threadIdx.x;                    // 8000 blocks x 512
  float x = (tok < TOKN) ? pitch[tok] : loud[tok - TOKN];
  float y = x * W0[c] + b0[c];
  __shared__ float rs[16], rs2[16];
  float s = y, s2 = y * y;
  for (int off = 16; off; off >>= 1) { s += __shfl_down(s, off); s2 += __shfl_down(s2, off); }
  int lane = c & 31, wv = c >> 5;
  if (lane == 0) { rs[wv] = s; rs2[wv] = s2; }
  __syncthreads();
  if (wv == 0) {
    float a = (lane < 16) ? rs[lane] : 0.f, a2 = (lane < 16) ? rs2[lane] : 0.f;
    for (int off = 8; off; off >>= 1) { a += __shfl_down(a, off); a2 += __shfl_down(a2, off); }
    if (lane == 0) { rs[0] = a; rs2[0] = a2; }
  }
  __syncthreads();
  float mu = rs[0] * (1.f / 512.f);
  float var = rs2[0] * (1.f / 512.f) - mu * mu;
  float v = (y - mu) * rsqrtf(var + 1e-5f) * g[c] + be[c];
  v = v > 0.f ? v : 0.01f * v;
  out[(size_t)tok * 512 + c] = (f16)v;
}

// ---------------- generic WMMA GEMM: Y = X(MxK,f16) * W^T(NxK,f16) + b ----------------
// doLN=1: LayerNorm(512)+leaky fused, f16 out (optionally split M / column offset)
__global__ void __launch_bounds__(256)
k_gemm(const f16* __restrict__ X, int ldx, int kTiles,
       const f16* __restrict__ W, int ldw,
       const float* __restrict__ bias,
       int nTiles, int colBase0,
       float* __restrict__ Yf, int ldyf,
       f16* __restrict__ Yh, int ldyh,
       int msplit, int splitColOff,
       const float* __restrict__ g, const float* __restrict__ be, int doLN) {
  __shared__ float smem[16 * 512];
  const int lane = threadIdx.x & 31;
  const int wave = threadIdx.x >> 5;
  const int m16  = lane & 15;
  const int half = lane >> 4;
  const int rowBase = blockIdx.x * 16;
  const int colBase = colBase0 + blockIdx.y * nTiles * 16;

  for (int tn = wave; tn < nTiles; tn += 8) {
    int colG = colBase + tn * 16 + m16;
    v8f c = {};
    const f16* xrow = X + (size_t)(rowBase + m16) * ldx;
    const f16* wrow = W + (size_t)colG * ldw;
    for (int kt = 0; kt < kTiles; ++kt) {
      int kk = kt * 32;
      V16 a, b;
      a.h[0] = *(const v8h*)(xrow + kk + half * 8);
      a.h[1] = *(const v8h*)(xrow + kk + half * 8 + 16);
      b.h[0] = *(const v8h*)(wrow + kk + half * 8);
      b.h[1] = *(const v8h*)(wrow + kk + half * 8 + 16);
      c = __builtin_amdgcn_wmma_f32_16x16x32_f16(false, a.v, false, b.v, (short)0, c, false, false);
    }
    float bv = bias ? bias[colG] : 0.f;
    if (!doLN) {
#pragma unroll
      for (int r = 0; r < 8; ++r) {
        int m = half * 8 + r;
        if (Yf) Yf[(size_t)(rowBase + m) * ldyf + colG] = c[r] + bv;
      }
    } else {
#pragma unroll
      for (int r = 0; r < 8; ++r)
        smem[(half * 8 + r) * 512 + tn * 16 + m16] = c[r] + bv;
    }
  }
  if (!doLN) return;
  __syncthreads();
  // LayerNorm over 512 + leaky relu; 8 waves x 2 rows
  for (int rr = 0; rr < 2; ++rr) {
    int m = wave * 2 + rr;
    const float* rowp = smem + m * 512;
    float s = 0.f, s2 = 0.f;
    for (int i = lane; i < 512; i += 32) { float v = rowp[i]; s += v; s2 += v * v; }
    for (int off = 16; off; off >>= 1) { s += __shfl_down(s, off); s2 += __shfl_down(s2, off); }
    s = __shfl(s, 0); s2 = __shfl(s2, 0);
    float mu = s * (1.f / 512.f);
    float var = s2 * (1.f / 512.f) - mu * mu;
    float rsg = rsqrtf(var + 1e-5f);
    int R = rowBase + m, orow = R, coff = 0;
    if (msplit > 0 && R >= msplit) { orow = R - msplit; coff = splitColOff; }
    for (int i = lane; i < 512; i += 32) {
      float v = (rowp[i] - mu) * rsg * g[i] + be[i];
      v = v > 0.f ? v : 0.01f * v;
      Yh[(size_t)orow * ldyh + coff + i] = (f16)v;
      if (Yf) Yf[(size_t)orow * ldyf + coff + i] = v;
    }
  }
}

// ---------------- GRU recurrence: 1 persistent workgroup, 32 waves ----------------
// gi slice for step t is DMA'd into LDS by the Tensor Data Mover, overlapped with
// the gh = h @ Whh^T WMMA computation; gates read it from LDS.
__global__ void __launch_bounds__(1024)
k_gru(const f16* __restrict__ Whh, const float* __restrict__ bhh,
      const float* __restrict__ gi, f16* __restrict__ yout) {
  __shared__ __align__(16) f16 h16[16 * 512];          // 16 KB (rows 4..15 zero)
  __shared__ float gh[4 * 1536];                       // 24 KB
  __shared__ __align__(16) float giBuf[4 * 1536];      // 24 KB (TDM destination)
  int tid = threadIdx.x;
  int lane = tid & 31, wave = tid >> 5;
  int m16 = lane & 15, half = lane >> 4;
  for (int i = tid; i < 16 * 512; i += 1024) h16[i] = (f16)0.f;
  __syncthreads();

  const unsigned ldsDst = (unsigned)(unsigned long long)&giBuf[0];

  for (int t = 0; t < FFR; ++t) {
    // ---- wave 0: issue TDM load of gi[:, t, :] (4 rows x 1536 f32) into LDS ----
    if (wave == 0) {
      unsigned long long ga = (unsigned long long)(gi + (size_t)t * 1536);
      u32x4 g0;
      g0[0] = 1u;                                       // count=1, user descriptor
      g0[1] = ldsDst;                                   // lds_addr
      g0[2] = (unsigned)(ga & 0xffffffffull);           // global_addr[31:0]
      g0[3] = (unsigned)((ga >> 32) & 0x01ffffffull)    // global_addr[56:32]
              | 0x80000000u;                            // type=2 (bits 127:126 = 10b)
      u32x8 g1;
      g1[0] = 0x00020000u;          // data_size=2 (4B), mask=0, no pad/iterate
      g1[1] = 0x06000000u;          // tensor_dim0 = 1536 (bits 63:48)
      g1[2] = 0x00040000u;          // tensor_dim1 = 4    (bits 95:80)
      g1[3] = 0x06000000u;          // tile_dim0 = 1536   (bits 127:112)
      g1[4] = 0x00000004u;          // tile_dim1 = 4, tile_dim2 = 0
      g1[5] = 1536000u;             // tensor_dim0_stride = F*1536 (elements)
      g1[6] = 0u;
      g1[7] = 0u;
      asm volatile("tensor_load_to_lds %0, %1" :: "s"(g0), "s"(g1) : "memory");
    }

    // ---- gh = h @ Whh^T : 96 tiles of 16 cols, 3 per wave (overlaps the DMA) ----
    for (int tt = 0; tt < 3; ++tt) {
      int tn = wave * 3 + tt;
      int colG = tn * 16 + m16;
      v8f c = {};
      const f16* wrow = Whh + (size_t)colG * 512;
      const f16* hrow = h16 + m16 * 512;
      for (int kt = 0; kt < 16; ++kt) {
        int kk = kt * 32;
        V16 a, b;
        a.h[0] = *(const v8h*)(hrow + kk + half * 8);
        a.h[1] = *(const v8h*)(hrow + kk + half * 8 + 16);
        b.h[0] = *(const v8h*)(wrow + kk + half * 8);
        b.h[1] = *(const v8h*)(wrow + kk + half * 8 + 16);
        c = __builtin_amdgcn_wmma_f32_16x16x32_f16(false, a.v, false, b.v, (short)0, c, false, false);
      }
#pragma unroll
      for (int r = 0; r < 8; ++r) {
        int m = half * 8 + r;
        if (m < 4) gh[m * 1536 + colG] = c[r];
      }
    }
    if (wave == 0) __builtin_amdgcn_s_wait_tensorcnt(0);   // DMA landed in LDS
    __syncthreads();

    // ---- gate nonlinearity + h update: 2048 elems / 1024 threads ----
    for (int e = tid; e < 2048; e += 1024) {
      int b = e >> 9, col = e & 511;
      size_t tok = (size_t)b * FFR + t;
      const float* gip = giBuf + b * 1536;
      float ir = gip[col],        hr = gh[b * 1536 + col]        + bhh[col];
      float iz = gip[512 + col],  hz = gh[b * 1536 + 512 + col]  + bhh[512 + col];
      float in = gip[1024 + col], hn = gh[b * 1536 + 1024 + col] + bhh[1024 + col];
      float r = sigf(ir + hr);
      float z = sigf(iz + hz);
      float n = tanhf(in + r * hn);
      float hp = (float)h16[b * 512 + col];
      float hv = (1.f - z) * n + z * hp;
      h16[b * 512 + col] = (f16)hv;
      yout[tok * 544 + col] = (f16)hv;   // mlp_out input, cols 0..511
    }
    __syncthreads();
  }
}

// fill mlp_out input cols 512 (pitch), 513 (loudness), 514..543 (zero pad)
__global__ void k_moin_fill(const float* __restrict__ pitch, const float* __restrict__ loud,
                            f16* __restrict__ moin) {
  int tok = blockIdx.x * 256 + threadIdx.x;
  if (tok >= TOKN) return;
  f16* p = moin + (size_t)tok * 544;
  p[512] = (f16)pitch[tok];
  p[513] = (f16)loud[tok];
#pragma unroll
  for (int i = 514; i < 544; ++i) p[i] = (f16)0.f;
}

// proj0 post: scale_function, anti-alias mask, normalize, * amplitude; phase delta
__global__ void __launch_bounds__(64)
k_scale0(const float* __restrict__ param0, const float* __restrict__ pitch,
         float* __restrict__ hd, float* __restrict__ dphi) {
  int fr = blockIdx.x, k = threadIdx.x;  // 4000 x 64
  __shared__ float s_amp, s_red[2];
  const float* row = param0 + (size_t)fr * 80;
  float p = pitch[fr];
  if (k == 0) {
    s_amp = scalef(row[0]);
    dphi[fr] = PI2F * p / 16000.f;
  }
  float s = scalef(row[1 + k]);
  float aa = ((p * (float)(k + 1) < 8000.f) ? 1.f : 0.f) + 1e-4f;
  s *= aa;
  float sum = s;
  for (int off = 16; off; off >>= 1) sum += __shfl_down(sum, off);
  if ((k & 31) == 0) s_red[k >> 5] = sum;
  __syncthreads();
  float tot = s_red[0] + s_red[1];
  hd[(size_t)fr * 64 + k] = s / tot * s_amp;
}

// per-batch serial phase prefix over frames
__global__ void k_pbase(const float* __restrict__ dphi, float* __restrict__ pbase) {
  int b = threadIdx.x;
  if (b >= NBATCH) return;
  float acc = 0.f;
  for (int f = 0; f < FFR; ++f) { pbase[b * FFR + f] = acc; acc += (float)BLK * dphi[b * FFR + f]; }
}

// harmonic additive synth
__global__ void k_harm(const float* __restrict__ pbase, const float* __restrict__ dphi,
                       const float* __restrict__ hd, float* __restrict__ sig) {
  long i = (long)blockIdx.x * 256 + threadIdx.x;
  if (i >= (long)NBATCH * TSAMP) return;
  int b = (int)(i / TSAMP), tl = (int)(i % TSAMP);
  int fr = tl / BLK, j = tl % BLK;
  int tok = b * FFR + fr;
  float phi = pbase[tok] + (float)(j + 1) * dphi[tok];
  phi -= PI2F * floorf(phi * (1.f / PI2F));   // mod 2pi (valid: integer harmonics)
  const float* h = hd + (size_t)tok * 64;
  float acc = 0.f;
#pragma unroll 8
  for (int k = 0; k < 64; ++k) acc += __sinf(phi * (float)(k + 1)) * h[k];
  sig[i] = acc;
}

// noise shaping: irfft(65)->128, roll/window/pad/roll -> 160-tap causal FIR, add to sig
__global__ void __launch_bounds__(160)
k_noise(const float* __restrict__ param1, const float* __restrict__ noise,
        float* __restrict__ sig) {
  int fr = blockIdx.x, j = threadIdx.x;  // 4000 x 160
  __shared__ float s2[65], wbuf[128], impl[160], nz[160];
  if (j < 65) s2[j] = scalef(param1[(size_t)fr * 80 + j] - 5.f);
  nz[j] = noise[(size_t)fr * 160 + j];
  __syncthreads();
  if (j < 128) {
    int n0 = (j + 64) & 127;               // roll by fs/2
    float c0 = PI2F * (float)n0 * (1.f / 128.f);
    float acc = s2[0] + s2[64] * __cosf(3.14159265f * (float)n0);
    for (int k = 1; k < 64; ++k) acc += 2.f * s2[k] * __cosf(c0 * (float)k);
    float win = 0.5f - 0.5f * __cosf(PI2F * (float)j * (1.f / 128.f));
    wbuf[j] = acc * (1.f / 128.f) * win;
  }
  __syncthreads();
  {
    int q = (j + 64) % 160;                // roll back by -fs/2 after pad to 160
    impl[j] = (q < 128) ? wbuf[q] : 0.f;
  }
  __syncthreads();
  float acc = 0.f;
  for (int q = 0; q <= j; ++q) acc += nz[q] * impl[j - q];
  int b = fr / FFR, fl = fr % FFR;
  sig[(size_t)b * TSAMP + fl * BLK + j] += acc;
}

// reverb impulse (16000 taps, tap0 forced to 1)
__global__ void k_imp(const float* __restrict__ rev_noise, const float* __restrict__ decay,
                      const float* __restrict__ wet, float* __restrict__ imp) {
  int t = blockIdx.x * 256 + threadIdx.x;
  if (t >= REVN) return;
  float d = decay[0], w = wet[0];
  float sp = log1pf(__expf(-d));                       // softplus(-decay)
  float env = __expf(-sp * ((float)t / 16000.f) * 500.f);
  float v = rev_noise[t] * env * sigf(w);
  imp[t] = (t == 0) ? 1.f : v;
}

// reverb: causal conv, LDS-tiled imp chunk + signal window
__global__ void __launch_bounds__(256)
k_reverb(const float* __restrict__ sig, const float* __restrict__ imp, float* __restrict__ out) {
  __shared__ float s_imp[256];
  __shared__ float s_sig[512];
  int b = blockIdx.y;
  int j0 = blockIdx.x * 256;
  int tid = threadIdx.x;
  const float* sb = sig + (size_t)b * TSAMP;
  float acc = 0.f;
  for (int d0 = 0; d0 < REVN; d0 += 256) {
    s_imp[tid] = imp[d0 + tid];
    if (d0 + 256 < REVN) __builtin_prefetch(imp + d0 + 256 + tid, 0, 0);
    int base = j0 - d0 - 255;
    for (int i = tid; i < 512; i += 256) {
      int gidx = base + i;
      s_sig[i] = (gidx >= 0 && gidx < TSAMP) ? sb[gidx] : 0.f;
    }
    __syncthreads();
#pragma unroll 8
    for (int dd = 0; dd < 256; ++dd) acc += s_imp[dd] * s_sig[tid + 255 - dd];
    __syncthreads();
  }
  out[(size_t)b * TSAMP + j0 + tid] = acc;
}

extern "C" void kernel_launch(void* const* d_in, const int* in_sizes, int n_in,
                              void* d_out, int out_size, void* d_ws, size_t ws_size,
                              hipStream_t stream) {
  // setup_inputs() insertion-order flattening:
  const float* pitch = (const float*)d_in[0];
  const float* loud  = (const float*)d_in[1];
  const float* noise = (const float*)d_in[2];
  // mlp_in: 3 layers x {W,b,g,beta} -> 3..14
  const float* Wi0 = (const float*)d_in[3],  *bi0 = (const float*)d_in[4];
  const float* gi0 = (const float*)d_in[5],  *be0 = (const float*)d_in[6];
  const float* Wi1 = (const float*)d_in[7],  *bi1 = (const float*)d_in[8];
  const float* gi1 = (const float*)d_in[9],  *be1 = (const float*)d_in[10];
  const float* Wi2 = (const float*)d_in[11], *bi2 = (const float*)d_in[12];
  const float* gi2 = (const float*)d_in[13], *be2 = (const float*)d_in[14];
  // gru
  const float* Wih = (const float*)d_in[15], *Whh = (const float*)d_in[16];
  const float* bih = (const float*)d_in[17], *bhh = (const float*)d_in[18];
  // mlp_out
  const float* Wo0 = (const float*)d_in[19], *bo0 = (const float*)d_in[20];
  const float* go0 = (const float*)d_in[21], *bt0 = (const float*)d_in[22];
  const float* Wo1 = (const float*)d_in[23], *bo1 = (const float*)d_in[24];
  const float* go1 = (const float*)d_in[25], *bt1 = (const float*)d_in[26];
  const float* Wo2 = (const float*)d_in[27], *bo2 = (const float*)d_in[28];
  const float* go2 = (const float*)d_in[29], *bt2 = (const float*)d_in[30];
  const float* P0W = (const float*)d_in[31], *P0b = (const float*)d_in[32];
  const float* P1W = (const float*)d_in[33], *P1b = (const float*)d_in[34];
  const float* revn = (const float*)d_in[35];
  const float* rdec = (const float*)d_in[36];
  const float* rwet = (const float*)d_in[37];

  // ---- workspace layout ----
  char* wp = (char*)d_ws;
  size_t off = 0;
  auto ALLOC = [&](size_t bytes) -> void* {
    void* r = wp + off; off = (off + bytes + 255) & ~(size_t)255; return r;
  };
  f16* W1in16 = (f16*)ALLOC(512 * 512 * 2);
  f16* W2in16 = (f16*)ALLOC(512 * 512 * 2);
  f16* Wih16  = (f16*)ALLOC(1536 * 1024 * 2);
  f16* Whh16  = (f16*)ALLOC(1536 * 512 * 2);
  f16* Wo016  = (f16*)ALLOC(512 * 544 * 2);
  f16* Wo116  = (f16*)ALLOC(512 * 512 * 2);
  f16* Wo216  = (f16*)ALLOC(512 * 512 * 2);
  f16* P016   = (f16*)ALLOC(80 * 512 * 2);
  f16* P116   = (f16*)ALLOC(80 * 512 * 2);
  float* pb0p = (float*)ALLOC(80 * 4);
  float* pb1p = (float*)ALLOC(80 * 4);
  f16* actA   = (f16*)ALLOC((size_t)8000 * 512 * 2);
  f16* actB   = (f16*)ALLOC((size_t)8000 * 512 * 2);
  f16* gruin  = (f16*)ALLOC((size_t)4000 * 1024 * 2);
  float* gibuf= (float*)ALLOC((size_t)4000 * 1536 * 4);
  f16* moin   = (f16*)ALLOC((size_t)4000 * 544 * 2);
  f16* actC   = (f16*)ALLOC((size_t)4000 * 512 * 2);
  f16* actD   = (f16*)ALLOC((size_t)4000 * 512 * 2);
  f16* actE   = (f16*)ALLOC((size_t)4000 * 512 * 2);
  float* par0 = (float*)ALLOC((size_t)4000 * 80 * 4);
  float* par1 = (float*)ALLOC((size_t)4000 * 80 * 4);
  float* hd   = (float*)ALLOC((size_t)4000 * 64 * 4);
  float* dphi = (float*)ALLOC(4000 * 4);
  float* pbase= (float*)ALLOC(4000 * 4);
  float* sig  = (float*)ALLOC((size_t)NBATCH * TSAMP * 4);
  float* imp  = (float*)ALLOC(REVN * 4);
  (void)ws_size; (void)n_in; (void)in_sizes; (void)out_size;

  auto CVT = [&](const float* s, f16* d, int N, int K, int Np, int Kp) {
    int n = Np * Kp;
    k_cvt<<<(n + 255) / 256, 256, 0, stream>>>(s, d, N, K, Np, Kp);
  };
  CVT(Wi1, W1in16, 512, 512, 512, 512);
  CVT(Wi2, W2in16, 512, 512, 512, 512);
  CVT(Wih, Wih16, 1536, 1024, 1536, 1024);
  CVT(Whh, Whh16, 1536, 512, 1536, 512);
  CVT(Wo0, Wo016, 512, 514, 512, 544);
  CVT(Wo1, Wo116, 512, 512, 512, 512);
  CVT(Wo2, Wo216, 512, 512, 512, 512);
  CVT(P0W, P016, 65, 512, 80, 512);
  CVT(P1W, P116, 65, 512, 80, 512);
  k_padbias<<<1, 256, 0, stream>>>(P0b, pb0p, 65, 80);
  k_padbias<<<1, 256, 0, stream>>>(P1b, pb1p, 65, 80);

  // mlp_in layer 1 on pitch(rows 0..3999) + loudness(rows 4000..7999)
  k_mlp_l1<<<8000, 512, 0, stream>>>(pitch, loud, Wi0, bi0, gi0, be0, actA);
  // mlp_in layer 2
  k_gemm<<<dim3(500, 1), 256, 0, stream>>>(actA, 512, 16, W1in16, 512, bi1,
      32, 0, nullptr, 0, actB, 512, 0, 0, gi1, be1, 1);
  // mlp_in layer 3 -> gru input [4000x1024] (pitch half cols 0..511, loudness cols 512..1023)
  k_gemm<<<dim3(500, 1), 256, 0, stream>>>(actB, 512, 16, W2in16, 512, bi2,
      32, 0, nullptr, 0, gruin, 1024, 4000, 512, gi2, be2, 1);
  // gi = X @ Wih^T + bih  (M=4000,K=1024,N=1536)
  k_gemm<<<dim3(250, 3), 256, 0, stream>>>(gruin, 1024, 32, Wih16, 1024, bih,
      32, 0, gibuf, 1536, nullptr, 0, 0, 0, nullptr, nullptr, 0);
  // GRU recurrence (writes moin cols 0..511); gi streamed in via TDM
  k_gru<<<1, 1024, 0, stream>>>(Whh16, bhh, gibuf, moin);
  k_moin_fill<<<16, 256, 0, stream>>>(pitch, loud, moin);
  // mlp_out
  k_gemm<<<dim3(250, 1), 256, 0, stream>>>(moin, 544, 17, Wo016, 544, bo0,
      32, 0, nullptr, 0, actC, 512, 0, 0, go0, bt0, 1);
  k_gemm<<<dim3(250, 1), 256, 0, stream>>>(actC, 512, 16, Wo116, 512, bo1,
      32, 0, nullptr, 0, actD, 512, 0, 0, go1, bt1, 1);
  k_gemm<<<dim3(250, 1), 256, 0, stream>>>(actD, 512, 16, Wo216, 512, bo2,
      32, 0, nullptr, 0, actE, 512, 0, 0, go2, bt2, 1);
  // projections
  k_gemm<<<dim3(250, 1), 256, 0, stream>>>(actE, 512, 16, P016, 512, pb0p,
      5, 0, par0, 80, nullptr, 0, 0, 0, nullptr, nullptr, 0);
  k_gemm<<<dim3(250, 1), 256, 0, stream>>>(actE, 512, 16, P116, 512, pb1p,
      5, 0, par1, 80, nullptr, 0, 0, 0, nullptr, nullptr, 0);
  // harmonic branch
  k_scale0<<<4000, 64, 0, stream>>>(par0, pitch, hd, dphi);
  k_pbase<<<1, 4, 0, stream>>>(dphi, pbase);
  k_harm<<<(NBATCH * TSAMP + 255) / 256, 256, 0, stream>>>(pbase, dphi, hd, sig);
  // noise branch (adds into sig)
  k_noise<<<4000, 160, 0, stream>>>(par1, noise, sig);
  // reverb
  k_imp<<<(REVN + 255) / 256, 256, 0, stream>>>(revn, rdec, rwet, imp);
  k_reverb<<<dim3(TSAMP / 256, NBATCH), 256, 0, stream>>>(sig, imp, (float*)d_out);
}
